// SparseDualPathAttentionTFF_82910048682142
// MI455X (gfx1250) — matrix operand order repaired and checked by reading, not compiled, not run on hardware
//
#include <hip/hip_runtime.h>
#include <math.h>

#define BB 2
#define NN 4096
#define CC 256
#define HH 8
#define KK 32
#define DD 32
#define EPSF 1e-5f

typedef unsigned short u16;
typedef __attribute__((ext_vector_type(16))) __bf16 v16bf;
typedef __attribute__((ext_vector_type(8)))  float  v8f;
typedef __attribute__((ext_vector_type(4)))  unsigned int ui32x4;
typedef __attribute__((ext_vector_type(8)))  int i32x8;
typedef __attribute__((ext_vector_type(4)))  int i32x4;

union FRAG { v16bf v; ui32x4 q[2]; };

__device__ __forceinline__ u16 f2bf(float f) {
  union { float f; unsigned u; } x; x.f = f;
  unsigned r = x.u + 0x7FFFu + ((x.u >> 16) & 1u);   // round-to-nearest-even
  return (u16)(r >> 16);
}

// ---------------------------------------------------------------------------
// TDM: 2-D bf16 tile load (global -> LDS). D# packed per CDNA5 ISA ch.8.
// group0: count=1 | lds_addr | global_addr | type=2
// group1: data_size=2B, tensor_dim0/1, tile_dim0/1, tensor_dim0_stride
// ---------------------------------------------------------------------------
__device__ __forceinline__ void tdm_load_2d_bf16(
    unsigned lds_off, const u16* gptr,
    unsigned tile0 /*contig elems*/, unsigned tile1 /*rows*/,
    unsigned tensor0 /*row len elems*/, unsigned long stride0 /*elems*/,
    unsigned tensor1 /*total rows*/)
{
  unsigned long ga = (unsigned long)(uintptr_t)gptr;
  ui32x4 g0;
  g0[0] = 1u;                                            // count=1, user mode
  g0[1] = lds_off;                                       // LDS byte address
  g0[2] = (unsigned)ga;                                  // global addr lo
  g0[3] = (unsigned)((ga >> 32) & 0x01FFFFFFu) | (2u << 30); // hi | type=2

  unsigned long q0 = (1ul << 16)                                   // data_size=2B
                   | ((unsigned long)(tensor0 & 0xFFFFu) << 48);   // td0[15:0]
  unsigned long q1 = ((unsigned long)(tensor0 >> 16) & 0xFFFFu)    // td0[31:16]
                   | ((unsigned long)tensor1 << 16)                // td1 @ [111:80]
                   | ((unsigned long)(tile0 & 0xFFFFu) << 48);     // tile0 @ [127:112]
  unsigned long q2 = (unsigned long)(tile1 & 0xFFFFu)              // tile1 @ [143:128]
                   | ((stride0 & 0xFFFFFFFFul) << 32);             // stride0 lo
  unsigned long q3 = ((stride0 >> 32) & 0xFFFFul);                 // stride0 hi (dim1_stride=0)
  i32x8 g1;
  g1[0] = (int)(unsigned)q0; g1[1] = (int)(q0 >> 32);
  g1[2] = (int)(unsigned)q1; g1[3] = (int)(q1 >> 32);
  g1[4] = (int)(unsigned)q2; g1[5] = (int)(q2 >> 32);
  g1[6] = (int)(unsigned)q3; g1[7] = (int)(q3 >> 32);
  i32x4 z4 = {};
#if __clang_major__ >= 23
  i32x8 z8 = {};
  __builtin_amdgcn_tensor_load_to_lds(g0, g1, z4, z4, z8, 0);
#else
  __builtin_amdgcn_tensor_load_to_lds(g0, g1, z4, z4, 0);
#endif
}

// ---------------------------------------------------------------------------
// Transpose (B,C,N)->(B,N,C); optional LayerNorm into fused slot (stride 3C).
// Dual-writes a bf16 copy of whatever feeds downstream GEMMs.
// ---------------------------------------------------------------------------
template<bool DO_LN>
__global__ __launch_bounds__(256) void transpose_ln_kernel(
    const float* __restrict__ feat, const float* __restrict__ lw,
    const float* __restrict__ lb, float* __restrict__ xout,
    float* __restrict__ nout, u16* __restrict__ bfout)
{
  __shared__ float tile[CC][33];
  __shared__ float ssum[8][32], ssq[8][32];
  __shared__ float smean[32], srstd[32];
  int tid = threadIdx.x;
  int b = blockIdx.y;
  int n0 = blockIdx.x * 32;
  int nloc = tid & 31, cg = tid >> 5;
  int n = n0 + nloc;

  float sum = 0.f, sq = 0.f;
  for (int c = cg * 32; c < cg * 32 + 32; ++c) {
    float v = feat[(b * CC + c) * NN + n];   // coalesced over nloc
    tile[c][nloc] = v; sum += v; sq += v * v;
  }
  ssum[cg][nloc] = sum; ssq[cg][nloc] = sq;
  __syncthreads();
  if (tid < 32) {
    float s = 0.f, q = 0.f;
    for (int g = 0; g < 8; ++g) { s += ssum[g][tid]; q += ssq[g][tid]; }
    float m = s * (1.f / CC);
    float var = q * (1.f / CC) - m * m;
    smean[tid] = m; srstd[tid] = rsqrtf(var + EPSF);
  }
  __syncthreads();
  for (int t = tid; t < 32 * CC; t += 256) {
    int ni = t >> 8, c = t & 255;
    float v = tile[c][ni];
    int rowg = b * NN + n0 + ni;
    xout[rowg * CC + c] = v;                 // coalesced over c
    if (DO_LN) {
      float nv = (v - smean[ni]) * srstd[ni] * lw[c] + lb[c];
      nout[(size_t)rowg * (3 * CC) + c] = nv;
      bfout[(size_t)rowg * (3 * CC) + c] = f2bf(nv);
    } else {
      bfout[(size_t)rowg * CC + c] = f2bf(v);
    }
  }
}

// ---------------------------------------------------------------------------
// Packed fp32 -> bf16 conversion (weights), 4 elems/thread
// ---------------------------------------------------------------------------
__global__ __launch_bounds__(256) void cvt_bf16_kernel(
    const float* __restrict__ src, u16* __restrict__ dst, int n4)
{
  int i = blockIdx.x * 256 + threadIdx.x;
  if (i >= n4) return;
  float4 v = ((const float4*)src)[i];
  uint2 p;
  p.x = (unsigned)f2bf(v.x) | ((unsigned)f2bf(v.y) << 16);
  p.y = (unsigned)f2bf(v.z) | ((unsigned)f2bf(v.w) << 16);
  ((uint2*)dst)[i] = p;
}

// ---------------------------------------------------------------------------
// KNN: one block per query; d2 row in LDS; 32x block-argmin extraction
// ---------------------------------------------------------------------------
__global__ __launch_bounds__(256) void knn_kernel(
    const float* __restrict__ xyzA, const float* __restrict__ xyzB,
    int* __restrict__ knn_idx)
{
  __shared__ float d2s[NN];
  __shared__ float rv[256];
  __shared__ int   ri[256];
  int tid = threadIdx.x;
  int b = blockIdx.y, n = blockIdx.x;
  float ax = xyzA[(b * NN + n) * 3 + 0];
  float ay = xyzA[(b * NN + n) * 3 + 1];
  float az = xyzA[(b * NN + n) * 3 + 2];
  for (int m = tid; m < NN; m += 256) {
    float dx = ax - xyzB[(b * NN + m) * 3 + 0];
    float dy = ay - xyzB[(b * NN + m) * 3 + 1];
    float dz = az - xyzB[(b * NN + m) * 3 + 2];
    d2s[m] = dx * dx + dy * dy + dz * dz;
  }
  __syncthreads();
  for (int sel = 0; sel < KK; ++sel) {
    float bv = 3.4e38f; int bi = 0x7FFFFFFF;
    for (int m = tid; m < NN; m += 256) {
      float v = d2s[m];
      if (v < bv) { bv = v; bi = m; }
    }
    rv[tid] = bv; ri[tid] = bi;
    __syncthreads();
    for (int s = 128; s > 0; s >>= 1) {
      if (tid < s) {
        float v2 = rv[tid + s]; int i2 = ri[tid + s];
        if (v2 < rv[tid] || (v2 == rv[tid] && i2 < ri[tid])) { rv[tid] = v2; ri[tid] = i2; }
      }
      __syncthreads();
    }
    if (tid == 0) {
      knn_idx[(b * NN + n) * KK + sel] = ri[0];
      d2s[ri[0]] = 3.4e38f;
    }
    __syncthreads();
  }
}

// ---------------------------------------------------------------------------
// GEMM out[m,o] = sum_c A[m,c]*W[o,c], A/W already bf16.
// Block tile 16(M) x 128(N); per K-step the TDM stages A(16x32) and B(128x32)
// into double-buffered LDS (wave 0 issues, pipelined one step ahead via
// s_wait_tensorcnt<=2); 8 waves each do one 16x16 WMMA from b128 LDS loads.
// EPI 0: +bias fp32 store. EPI 1: BN(eval)+ReLU -> bf16 store.
// EPI 2: +bias+residual -> transposed (B,C,N) fp32 store.
// ---------------------------------------------------------------------------
template<int EPI>
__global__ __launch_bounds__(256) void gemm_wmma_tdm_kernel(
    const u16* __restrict__ A, int lda,
    const u16* __restrict__ W, int Kdim,
    const float* __restrict__ e0, const float* __restrict__ e1,
    void* out, int Nout, int ngroups)
{
  __shared__ __align__(16) u16 sA[2][16 * 32];     // 2 x 1KB
  __shared__ __align__(16) u16 sB[2][128 * 32];    // 2 x 8KB

  int lane = threadIdx.x & 31;
  int wave = threadIdx.x >> 5;
  int mtile = blockIdx.x / ngroups;
  int gn = blockIdx.x % ngroups;
  int row0 = mtile * 16;
  int col0g = gn * 128;
  int ksteps = Kdim >> 5;

  // fragment coords (wave32 WMMA; contiguous k order: lane half h holds k 16h..16h+15)
  int ar  = lane & 15;            // A row
  int akb = (lane >> 4) * 16;     // A k-base: 0 / 16
  int br  = wave * 16 + (lane & 15); // B row in LDS tile (= output col)
  int bkb = (lane >> 4) * 16;     // B k-base: 0 / 16

  unsigned ldsA[2] = { (unsigned)(uintptr_t)&sA[0][0], (unsigned)(uintptr_t)&sA[1][0] };
  unsigned ldsB[2] = { (unsigned)(uintptr_t)&sB[0][0], (unsigned)(uintptr_t)&sB[1][0] };

  if (wave == 0) {  // prologue: stage k-step 0
    tdm_load_2d_bf16(ldsA[0], A + (size_t)row0 * lda, 32, 16,
                     (unsigned)Kdim, (unsigned long)lda, 1u << 30);
    tdm_load_2d_bf16(ldsB[0], W + (size_t)col0g * Kdim, 32, 128,
                     (unsigned)Kdim, (unsigned long)Kdim, 1u << 30);
  }

  v8f acc = {};
  for (int ks = 0; ks < ksteps; ++ks) {
    int cur = ks & 1;
    if (wave == 0) {
      if (ks + 1 < ksteps) {
        int k1 = (ks + 1) * 32;
        tdm_load_2d_bf16(ldsA[cur ^ 1], A + (size_t)row0 * lda + k1, 32, 16,
                         (unsigned)Kdim, (unsigned long)lda, 1u << 30);
        tdm_load_2d_bf16(ldsB[cur ^ 1], W + (size_t)col0g * Kdim + k1, 32, 128,
                         (unsigned)Kdim, (unsigned long)Kdim, 1u << 30);
        __builtin_amdgcn_s_wait_tensorcnt((short)2);  // pair(ks) complete
      } else {
        __builtin_amdgcn_s_wait_tensorcnt((short)0);
      }
    }
    __syncthreads();                    // buffers[cur] visible to all waves

    FRAG fa, fb;
    const u16* ap = sA[cur] + ar * 32;
    fa.q[0] = *reinterpret_cast<const ui32x4*>(ap + akb);        // K akb..akb+7
    fa.q[1] = *reinterpret_cast<const ui32x4*>(ap + akb + 8);    // K akb+8..akb+15
    const u16* bp = sB[cur] + br * 32;
    fb.q[0] = *reinterpret_cast<const ui32x4*>(bp + bkb);        // K bkb..bkb+7
    fb.q[1] = *reinterpret_cast<const ui32x4*>(bp + bkb + 8);    // K bkb+8..bkb+15
    acc = __builtin_amdgcn_wmma_f32_16x16x32_bf16(
        false, fa.v, false, fb.v, (short)0, acc, false, false);

    __syncthreads();                    // done reading before TDM reuses buffer
  }

  // C/D layout: acc[r] -> M = r + 8*(lane>=16), N = lane%16
  int col = col0g + wave * 16 + (lane & 15);
  int rbase = row0 + (lane >> 4) * 8;
  if (EPI == 0) {
    float* o = (float*)out;
    float bias = e0[col];
    #pragma unroll
    for (int r = 0; r < 8; ++r) o[(size_t)(rbase + r) * Nout + col] = acc[r] + bias;
  } else if (EPI == 1) {
    u16* o = (u16*)out;
    float scale = e0[col] * rsqrtf(1.f + EPSF);   // BN eval: mean 0, var 1
    float shift = e1[col];
    #pragma unroll
    for (int r = 0; r < 8; ++r) {
      float t = acc[r] * scale + shift;
      o[(size_t)(rbase + r) * Nout + col] = f2bf(t > 0.f ? t : 0.f);
    }
  } else {
    float* o = (float*)out;
    float bias = e0[col];
    #pragma unroll
    for (int r = 0; r < 8; ++r) {
      int row = rbase + r;
      float valr = acc[r] + bias + e1[(size_t)row * CC + col];  // residual xA
      int bb2 = row >> 12, nl = row & (NN - 1);
      o[((size_t)bb2 * CC + col) * NN + nl] = valr;             // (B,C,N)
    }
  }
}

// ---------------------------------------------------------------------------
// Dual-path attention: one wave per (b,n,h); lane = neighbor k (K==wave32).
// ---------------------------------------------------------------------------
__global__ __launch_bounds__(256) void attn_kernel(
    const int* __restrict__ knn_idx,
    const float* qs, const float* __restrict__ ks,
    const float* __restrict__ vv, const float* qd,
    const float* __restrict__ kd,
    float* ctxs, float* ctxd)
{
  int lane = threadIdx.x & 31;
  int wave = threadIdx.x >> 5;
  int gw = blockIdx.x * 8 + wave;          // gw = ((b*NN + n)*HH + h)
  int h = gw & (HH - 1);
  int n = (gw >> 3) & (NN - 1);
  int b = gw >> 15;
  int row = b * NN + n;
  int hb = h * DD;

  int idx  = knn_idx[row * KK + lane];
  int krow = b * NN + idx;

  float qv  = qs[(size_t)row * CC + hb + lane];
  float qdv = qd[(size_t)row * CC + hb + lane];

  float sim = 0.f, dsq = 0.f;
  #pragma unroll 8
  for (int d = 0; d < DD; ++d) {
    float kv1 = ks[(size_t)krow * CC + hb + d];
    float kv2 = kd[(size_t)krow * CC + hb + d];
    sim += __shfl(qv, d, 32) * kv1;
    float df = __shfl(qdv, d, 32) - kv2;
    dsq += df * df;
  }
  sim *= 0.17677669529663687f;   // 1/sqrt(D)
  float dist = sqrtf(dsq);

  float m1 = sim, m2 = dist;
  for (int o = 16; o > 0; o >>= 1) {
    m1 = fmaxf(m1, __shfl_xor(m1, o, 32));
    m2 = fmaxf(m2, __shfl_xor(m2, o, 32));
  }
  float e1v = expf(sim - m1), e2v = expf(dist - m2);
  float s1 = e1v, s2 = e2v;
  for (int o = 16; o > 0; o >>= 1) {
    s1 += __shfl_xor(s1, o, 32);
    s2 += __shfl_xor(s2, o, 32);
  }
  float w1 = e1v / s1, w2 = e2v / s2;

  float cs = 0.f, cd = 0.f;
  #pragma unroll 8
  for (int k2 = 0; k2 < KK; ++k2) {
    float wks = __shfl(w1, k2, 32);
    float wkd = __shfl(w2, k2, 32);
    int   kr  = __shfl(krow, k2, 32);
    float vval = vv[(size_t)kr * CC + hb + lane];  // coalesced over lane
    cs += wks * vval; cd += wkd * vval;
  }
  ctxs[(size_t)row * CC + hb + lane] = cs;
  ctxd[(size_t)row * CC + hb + lane] = cd;
}

// ---------------------------------------------------------------------------
// Row LayerNorm (C=256) -> fp32 dst + bf16 dst (both stride `dstride`)
// ---------------------------------------------------------------------------
__global__ __launch_bounds__(256) void ln_rows_kernel(
    const float* __restrict__ src, const float* __restrict__ lw,
    const float* __restrict__ lb, float* __restrict__ dst,
    u16* __restrict__ bdst, int dstride)
{
  __shared__ float s1[256], s2[256];
  int tid = threadIdx.x;
  int row = blockIdx.x;
  float v = src[(size_t)row * CC + tid];
  s1[tid] = v; s2[tid] = v * v;
  __syncthreads();
  for (int s = 128; s > 0; s >>= 1) {
    if (tid < s) { s1[tid] += s1[tid + s]; s2[tid] += s2[tid + s]; }
    __syncthreads();
  }
  float m = s1[0] * (1.f / CC);
  float var = s2[0] * (1.f / CC) - m * m;
  float rstd = rsqrtf(var + EPSF);
  float res = (v - m) * rstd * lw[tid] + lb[tid];
  dst[(size_t)row * dstride + tid] = res;
  bdst[(size_t)row * dstride + tid] = f2bf(res);
}

// ---------------------------------------------------------------------------
extern "C" void kernel_launch(void* const* d_in, const int* in_sizes, int n_in,
                              void* d_out, int out_size, void* d_ws, size_t ws_size,
                              hipStream_t stream)
{
  const float* xyzA   = (const float*)d_in[0];
  const float* xyzB   = (const float*)d_in[1];
  const float* featA  = (const float*)d_in[2];
  const float* featB  = (const float*)d_in[3];
  const float* ln_in_w = (const float*)d_in[4];
  const float* ln_in_b = (const float*)d_in[5];
  const float* wq  = (const float*)d_in[6];   const float* bq  = (const float*)d_in[7];
  const float* wk  = (const float*)d_in[8];   const float* bk  = (const float*)d_in[9];
  const float* wv  = (const float*)d_in[10];  const float* bv  = (const float*)d_in[11];
  const float* wqd = (const float*)d_in[12];  const float* bqd = (const float*)d_in[13];
  const float* wkd = (const float*)d_in[14];  const float* bkd = (const float*)d_in[15];
  const float* ln_sim_w = (const float*)d_in[16]; const float* ln_sim_b = (const float*)d_in[17];
  const float* ln_dis_w = (const float*)d_in[18]; const float* ln_dis_b = (const float*)d_in[19];
  const float* fus_w1 = (const float*)d_in[20];
  const float* bn_g   = (const float*)d_in[21]; const float* bn_b = (const float*)d_in[22];
  const float* fus_w2 = (const float*)d_in[23]; const float* fus_b2 = (const float*)d_in[24];
  float* out = (float*)d_out;

  float* ws = (float*)d_ws;
  size_t off = 0;
  const size_t MR = (size_t)BB * NN;                 // 8192 rows
  float* xA    = ws + off; off += MR * CC;           // raw A (residual)
  float* xB    = ws + off; off += MR * CC;           // raw B
  float* fused = ws + off; off += MR * 3 * CC;       // [xA_n | ln(ctx_s) | ln(ctx_d)] fp32
  float* q_s   = ws + off; off += MR * CC;           // reused as ctx_s raw
  float* k_s   = ws + off; off += MR * CC;
  float* v_b   = ws + off; off += MR * CC;
  float* q_d   = ws + off; off += MR * CC;           // reused as ctx_d raw
  float* k_d   = ws + off; off += MR * CC;
  int* knn_idx = (int*)(ws + off); off += MR * KK;
  // bf16 regions (counted in float slots, 2 u16 per float)
  u16* fused_bf = (u16*)(ws + off); off += MR * 3 * CC / 2;
  u16* xB_bf    = (u16*)(ws + off); off += MR * CC / 2;
  u16* y1_bf    = (u16*)(ws + off); off += MR * 2 * CC / 2;
  u16* wq_bf    = (u16*)(ws + off); off += (CC * CC) / 2;
  u16* wk_bf    = (u16*)(ws + off); off += (CC * CC) / 2;
  u16* wv_bf    = (u16*)(ws + off); off += (CC * CC) / 2;
  u16* wqd_bf   = (u16*)(ws + off); off += (CC * CC) / 2;
  u16* wkd_bf   = (u16*)(ws + off); off += (CC * CC) / 2;
  u16* w1_bf    = (u16*)(ws + off); off += (2 * CC * 3 * CC) / 2;
  u16* w2_bf    = (u16*)(ws + off); off += (CC * 2 * CC) / 2;

  dim3 blk(256);

  // one-time weight conversions (fp32 -> packed bf16)
  auto cvt = [&](const float* s, u16* d, int n) {
    cvt_bf16_kernel<<<(n / 4 + 255) / 256, blk, 0, stream>>>(s, d, n / 4);
  };
  cvt(wq,  wq_bf,  CC * CC);  cvt(wk,  wk_bf,  CC * CC);  cvt(wv, wv_bf, CC * CC);
  cvt(wqd, wqd_bf, CC * CC);  cvt(wkd, wkd_bf, CC * CC);
  cvt(fus_w1, w1_bf, 2 * CC * 3 * CC);
  cvt(fus_w2, w2_bf, CC * 2 * CC);

  transpose_ln_kernel<true ><<<dim3(NN / 32, BB), blk, 0, stream>>>(
      featA, ln_in_w, ln_in_b, xA, fused, fused_bf);
  transpose_ln_kernel<false><<<dim3(NN / 32, BB), blk, 0, stream>>>(
      featB, ln_in_w, ln_in_b, xB, fused, xB_bf);

  knn_kernel<<<dim3(NN, BB), blk, 0, stream>>>(xyzA, xyzB, knn_idx);

  // projections: M=8192, Nout=256 -> 512 mtiles x 2 col-groups
  int projBlocks = (int)(MR / 16) * 2;               // 1024
  gemm_wmma_tdm_kernel<0><<<projBlocks, blk, 0, stream>>>(fused_bf, 3 * CC, wq_bf,  CC, bq,  nullptr, q_s, CC, 2);
  gemm_wmma_tdm_kernel<0><<<projBlocks, blk, 0, stream>>>(xB_bf,    CC,     wk_bf,  CC, bk,  nullptr, k_s, CC, 2);
  gemm_wmma_tdm_kernel<0><<<projBlocks, blk, 0, stream>>>(xB_bf,    CC,     wv_bf,  CC, bv,  nullptr, v_b, CC, 2);
  gemm_wmma_tdm_kernel<0><<<projBlocks, blk, 0, stream>>>(fused_bf, 3 * CC, wqd_bf, CC, bqd, nullptr, q_d, CC, 2);
  gemm_wmma_tdm_kernel<0><<<projBlocks, blk, 0, stream>>>(xB_bf,    CC,     wkd_bf, CC, bkd, nullptr, k_d, CC, 2);

  attn_kernel<<<(int)(MR * HH / 8), blk, 0, stream>>>(knn_idx, q_s, k_s, v_b, q_d, k_d, q_s, q_d);

  ln_rows_kernel<<<(int)MR, blk, 0, stream>>>(q_s, ln_sim_w, ln_sim_b,
                                              fused + CC, fused_bf + CC, 3 * CC);
  ln_rows_kernel<<<(int)MR, blk, 0, stream>>>(q_d, ln_dis_w, ln_dis_b,
                                              fused + 2 * CC, fused_bf + 2 * CC, 3 * CC);

  // fusion MLP: GEMM1 (K=768, Nout=512, bf16 out), GEMM2 (K=512, Nout=256)
  int g1Blocks = (int)(MR / 16) * 4;                 // 2048
  gemm_wmma_tdm_kernel<1><<<g1Blocks, blk, 0, stream>>>(fused_bf, 3 * CC, w1_bf, 3 * CC, bn_g, bn_b, y1_bf, 2 * CC, 4);
  int g2Blocks = (int)(MR / 16) * 2;                 // 1024
  gemm_wmma_tdm_kernel<2><<<g2Blocks, blk, 0, stream>>>(y1_bf, 2 * CC, w2_bf, 2 * CC, fus_b2, xA, out, CC, 2);
}
